// UnifiedDecoder_52166672777978
// MI455X (gfx1250) — compile-verified, matrix operand order checked by reading, not run on hardware
//
#include <hip/hip_runtime.h>
#include <hip/hip_bf16.h>
#include <math.h>

// ---------------------------------------------------------------------------
// CDNA5 (gfx1250) UnifiedDecoder: persistent single-WGP recurrent kernel.
// Core GEMMs use V_WMMA_F32_16X16X32_F16 (f16 A/B, f32 accum).
// Round 2: restructured GRU inner step so only ONE WMMA accumulator set is
// live at a time (was zacc+pacc concurrently) -> removes VGPR spills seen as
// "scratch_load_b32 ... Folded Reload" in round-1 asm.
// ---------------------------------------------------------------------------

typedef __attribute__((ext_vector_type(16))) _Float16 v16h;
typedef __attribute__((ext_vector_type(8)))  float    v8f;

#define T_STEPS 64
#define B_SZ    64
#define L_SEG   16
#define HID     512
#define LAT     256
#define NNUC    5
#define NHPN    4
#define MAXNB   10
#define NSLOTS  (B_SZ * (T_STEPS + 1))   // 4160

// ---------------------------------------------------------------------------
// Weight repack: W[n][k] fp32 -> f16 B-operand fragments.
// Layout per CDNA5 ISA 7.12.2 (16-bit A/B 16x32 fragment): lane L holds
// n = nt*16 + (L&15); vgpr v holds K pair k = chunk*32 + (L>=16?8:0)
// + (v&3)*2 + (v>>2)*16.  Flat pair index: (((nt*Kc + c)*32 + lane)*8 + v).
// ---------------------------------------------------------------------------
__global__ void pack_kernel(const float* __restrict__ W, int ldw, int colOff,
                            int Kc, _Float16* __restrict__ dst) {
  int idx = blockIdx.x * blockDim.x + threadIdx.x;
  int total = 32 * Kc * 32 * 8;
  if (idx >= total) return;
  int v    = idx & 7;
  int lane = (idx >> 3) & 31;
  int cc   = (idx >> 8) % Kc;
  int nt   = (idx >> 8) / Kc;
  int n = nt * 16 + (lane & 15);
  int k = cc * 32 + ((lane >> 4) << 3) + ((v & 3) << 1) + ((v >> 2) << 4);
  const float* row = W + (size_t)n * ldw + colOff;
  dst[(size_t)idx * 2]     = (_Float16)row[k];
  dst[(size_t)idx * 2 + 1] = (_Float16)row[k + 1];
}

// -------------------------- device helpers ---------------------------------
__device__ inline v16h a_frag_f32(const float* A, int ld, int mt, int chunk, int lane) {
  const float* row = A + (size_t)(mt * 16 + (lane & 15)) * ld
                       + chunk * 32 + ((lane >> 4) << 3);
  v16h a;
#pragma unroll
  for (int v = 0; v < 8; ++v) {
    int k = ((v & 3) << 1) + ((v >> 2) << 4);
    a[2 * v]     = (_Float16)row[k];
    a[2 * v + 1] = (_Float16)row[k + 1];
  }
  return a;
}

__device__ inline v16h a_frag_f16(const _Float16* A, int ld, int mt, int chunk, int lane) {
  const _Float16* row = A + (size_t)(mt * 16 + (lane & 15)) * ld
                          + chunk * 32 + ((lane >> 4) << 3);
  v16h a;
#pragma unroll
  for (int v = 0; v < 8; ++v) {
    int k = ((v & 3) << 1) + ((v >> 2) << 4);
    a[2 * v]     = row[k];
    a[2 * v + 1] = row[k + 1];
  }
  return a;
}

__device__ inline void wmma4(v8f acc[4], v16h af, const _Float16* Bp,
                             int Kc, int c, int ng, int lane) {
#pragma unroll
  for (int j = 0; j < 4; ++j) {
    int nt = ng * 4 + j;
    v16h bf = *(const v16h*)(Bp + ((((size_t)(nt * Kc + c)) * 32 + lane) << 4));
    acc[j] = __builtin_amdgcn_wmma_f32_16x16x32_f16(
        false, af, false, bf, (short)0, acc[j], false, false);
  }
}

// GEMM over Kc chunks of fp32 LDS A (ld=512)
__device__ inline void gemm_ldsA(v8f acc[4], const float* A, const _Float16* Bp,
                                 int Kc, int mt, int ng, int lane) {
  for (int c = 0; c < Kc; ++c) {
    v16h af = a_frag_f32(A, HID, mt, c, lane);
    wmma4(acc, af, Bp, Kc, c, ng, lane);
  }
}

// GRU gate tail: bias + 5-dim x input (fp32 VALU), columns 0..4 of W[512][517]
__device__ inline void gate_tail(v8f acc[4], const float* W, const float* bias,
                                 const float* xb, int mt, int ng, int lane) {
#pragma unroll
  for (int j = 0; j < 4; ++j) {
    int n = (ng * 4 + j) * 16 + (lane & 15);
    const float* wr = W + (size_t)n * 517;
    float base = bias[n];
#pragma unroll
    for (int r = 0; r < 8; ++r) {
      int m = mt * 16 + r + ((lane >> 4) << 3);
      float s = base;
#pragma unroll
      for (int q = 0; q < NNUC; ++q) s += xb[m * 8 + q] * wr[q];
      acc[j][r] += s;
    }
  }
}

__device__ inline void bias_relu(v8f acc[4], const float* bias, int ng, int lane) {
#pragma unroll
  for (int j = 0; j < 4; ++j) {
    int n = (ng * 4 + j) * 16 + (lane & 15);
    float bv = bias[n];
#pragma unroll
    for (int r = 0; r < 8; ++r) {
      float v = acc[j][r] + bv;
      acc[j][r] = v > 0.f ? v : 0.f;
    }
  }
}

__device__ inline void apply_sigmoid(v8f acc[4]) {
#pragma unroll
  for (int j = 0; j < 4; ++j)
#pragma unroll
    for (int r = 0; r < 8; ++r)
      acc[j][r] = 1.f / (1.f + expf(-acc[j][r]));
}

__device__ inline void apply_tanh(v8f acc[4]) {
#pragma unroll
  for (int j = 0; j < 4; ++j)
#pragma unroll
    for (int r = 0; r < 8; ++r)
      acc[j][r] = tanhf(acc[j][r]);
}

// Store wave's 4 tiles (C layout: vgpr r, lane -> m = r + 8*(lane>=16), n = lane&15)
__device__ inline void store_tiles(float* Out, const v8f acc[4], int mt, int ng, int lane) {
#pragma unroll
  for (int j = 0; j < 4; ++j) {
    int n = (ng * 4 + j) * 16 + (lane & 15);
#pragma unroll
    for (int r = 0; r < 8; ++r) {
      int m = mt * 16 + r + ((lane >> 4) << 3);
      Out[m * HID + n] = acc[j][r];
    }
  }
}

// hpn-style head: g = relu(hsrc @ Wnl^T + bnl); logits = g @ Whead^T + bhead;
// returns this thread's CE(sum) contribution (nonzero for tid<64).
__device__ float hpn_head_loss(const float* hsrc, float* gtmp, float* logitsS,
                               const _Float16* pWnl, const float* bnl,
                               const float* Whead, const float* bhead,
                               const int* tgt, int tid, int lane, int mt, int ng) {
  v8f acc[4] = {};
  gemm_ldsA(acc, hsrc, pWnl, 16, mt, ng, lane);
  bias_relu(acc, bnl, ng, lane);
  __syncthreads();
  store_tiles(gtmp, acc, mt, ng, lane);
  __syncthreads();
  if (tid < B_SZ * NHPN) {
    int b = tid >> 2, c = tid & 3;
    float s = bhead[c];
    const float* g = gtmp + b * HID;
    const float* w = Whead + (size_t)c * HID;
    for (int k = 0; k < HID; ++k) s += g[k] * w[k];
    logitsS[b * 8 + c] = s;
  }
  __syncthreads();
  float l = 0.f;
  if (tid < B_SZ) {
    float mx = logitsS[tid * 8];
    for (int c = 1; c < NHPN; ++c) mx = fmaxf(mx, logitsS[tid * 8 + c]);
    float se = 0.f;
    for (int c = 0; c < NHPN; ++c) se += expf(logitsS[tid * 8 + c] - mx);
    l = logf(se) + mx - logitsS[tid * 8 + tgt[tid]];
  }
  __syncthreads();
  return l;
}

// ---------------------------------------------------------------------------
// Persistent decoder kernel: one workgroup, 1024 threads = 32 wave32 waves.
// ---------------------------------------------------------------------------
__global__ __launch_bounds__(1024) void decoder_kernel(
    const float* __restrict__ seq,        // [T,B,L,5]
    const float* __restrict__ tree,       // [B,256]
    const float* __restrict__ graph,      // [B,256]
    const int* __restrict__ msg_idx,      // [T,B,10]
    const int* __restrict__ msg_mask,     // [T,B,10]
    const int* __restrict__ widx,         // [T,B]
    const int* __restrict__ nuc_tgt,      // [T,B,L]
    const int* __restrict__ hpn_tgt,      // [T,B]
    const int* __restrict__ stop_tgt,     // [T,B]
    const int* __restrict__ hpn_lbl,      // [T,B]
    const int* __restrict__ root_tgt,     // [B]
    const float* Wz, const float* bz,
    const float* Wr, const float* br,
    const float* Wh, const float* bh,
    const float* Whpn, const float* bhpn,
    const float* Whpnnl, const float* bhpnnl,
    const float* Wnuc, const float* bnuc,
    const float* Wnucnl, const float* bnucnl,
    const float* Wtopo, const float* btopo,
    const float* Wtoponl, const float* btoponl,
    const _Float16* pWz, const _Float16* pWr, const _Float16* pWh,
    const _Float16* pWnucnl, const _Float16* pWhpnnl, const _Float16* pWtoponl,
    float* hbuf, float* out) {

  __shared__ float    hcur[B_SZ * HID];     // 128 KB : recurrent hidden
  __shared__ float    tmp [B_SZ * HID];     // 128 KB : r / r*h / pre / head acts
  __shared__ _Float16 glat[B_SZ * LAT];     //  32 KB : graph latent (f16, A operand)
  __shared__ float    xbuf[B_SZ * 8];       // staged x (5 used / row)
  __shared__ float    logitsS[B_SZ * 8];
  __shared__ float    red[1024];

  const int tid  = threadIdx.x;
  const int lane = tid & 31;
  const int wave = tid >> 5;
  const int mt   = wave >> 3;   // M tile (batch rows 16*mt..)
  const int ng   = wave & 7;    // N tile group (4 tiles)
  float loss = 0.f;

  // stage graph latent as f16
  for (int i = tid; i < B_SZ * LAT; i += 1024) glat[i] = (_Float16)graph[i];

  // root hidden = [tree_latent, zeros]; also seed hbuf slots 0..63
  for (int i = tid; i < B_SZ * HID; i += 1024) {
    int b = i >> 9, c = i & 511;
    float v = (c < LAT) ? tree[b * LAT + c] : 0.f;
    hcur[i] = v;
    hbuf[i] = v;
  }
  __syncthreads();

  // initial hpn loss on root hidden
  loss += hpn_head_loss(hcur, tmp, logitsS, pWhpnnl, bhpnnl,
                        Whpn, bhpn, root_tgt, tid, lane, mt, ng);

  for (int t = 0; t < T_STEPS; ++t) {
    // ---- gather messages: hcur = sum_nb mask * hbuf[idx] ----
    for (int i = tid; i < B_SZ * HID; i += 1024) {
      int b = i >> 9, c = i & 511;
      const int* mi = msg_idx + (t * B_SZ + b) * MAXNB;
      const int* mm = msg_mask + (t * B_SZ + b) * MAXNB;
      float s = 0.f;
      for (int nb = 0; nb < MAXNB; ++nb)
        s += (float)mm[nb] * hbuf[(size_t)mi[nb] * HID + c];
      hcur[i] = s;
    }
    __syncthreads();

    // ---- GRU over L nucleotide positions ----
    for (int l = 0; l < L_SEG; ++l) {
      if (tid < B_SZ * NNUC) {
        int b = tid / NNUC, q = tid % NNUC;
        xbuf[b * 8 + q] = seq[((size_t)(t * B_SZ + b) * L_SEG + l) * NNUC + q];
      }
      __syncthreads();

      // r gate -> tmp
      {
        v8f acc[4] = {};
        gemm_ldsA(acc, hcur, pWr, 16, mt, ng, lane);
        gate_tail(acc, Wr, br, xbuf, mt, ng, lane);
        apply_sigmoid(acc);
        store_tiles(tmp, acc, mt, ng, lane);
      }
      __syncthreads();
      // tmp = r * h
      for (int i = tid; i < B_SZ * HID; i += 1024) tmp[i] *= hcur[i];
      __syncthreads();

      // candidate gate (A = r*h in tmp) -> registers, then into tmp
      {
        v8f acc[4] = {};
        gemm_ldsA(acc, tmp, pWh, 16, mt, ng, lane);
        gate_tail(acc, Wh, bh, xbuf, mt, ng, lane);
        apply_tanh(acc);
        __syncthreads();                     // all reads of tmp (r*h) done
        store_tiles(tmp, acc, mt, ng, lane); // tmp now holds pre (candidate)
      }
      // z gate (A = hcur) -> registers only; tmp stores above don't conflict
      {
        v8f zacc[4] = {};
        gemm_ldsA(zacc, hcur, pWz, 16, mt, ng, lane);
        gate_tail(zacc, Wz, bz, xbuf, mt, ng, lane);
        apply_sigmoid(zacc);
        __syncthreads();   // all z-reads of hcur done; all pre stores visible

        // in-place blend (exclusive tile ownership of hcur; pre from tmp)
#pragma unroll
        for (int j = 0; j < 4; ++j) {
          int n = (ng * 4 + j) * 16 + (lane & 15);
#pragma unroll
          for (int r = 0; r < 8; ++r) {
            int m = mt * 16 + r + ((lane >> 4) << 3);
            float hold = hcur[m * HID + n];
            float zv = zacc[j][r];
            hcur[m * HID + n] = (1.f - zv) * hold + zv * tmp[m * HID + n];
          }
        }
      }
      __syncthreads();

      // ---- nuc head: g = relu([h, glat] @ Wnucnl^T + b) ----
      {
        v8f acc[4] = {};
        for (int c = 0; c < 24; ++c) {
          v16h af = (c < 16) ? a_frag_f32(hcur, HID, mt, c, lane)
                             : a_frag_f16(glat, LAT, mt, c - 16, lane);
          wmma4(acc, af, pWnucnl, 24, c, ng, lane);
        }
        bias_relu(acc, bnucnl, ng, lane);
        __syncthreads();                 // blend reads of tmp done everywhere
        store_tiles(tmp, acc, mt, ng, lane);
      }
      __syncthreads();
      if (tid < B_SZ * NNUC) {
        int b = tid / NNUC, c = tid % NNUC;
        float s = bnuc[c];
        const float* g = tmp + b * HID;
        const float* w = Wnuc + (size_t)c * HID;
        for (int k = 0; k < HID; ++k) s += g[k] * w[k];
        logitsS[b * 8 + c] = s;
      }
      __syncthreads();
      if (tid < B_SZ) {
        float mx = logitsS[tid * 8];
        for (int c = 1; c < NNUC; ++c) mx = fmaxf(mx, logitsS[tid * 8 + c]);
        float se = 0.f;
        for (int c = 0; c < NNUC; ++c) se += expf(logitsS[tid * 8 + c] - mx);
        int tg = nuc_tgt[(t * B_SZ + tid) * L_SEG + l];
        loss += logf(se) + mx - logitsS[tid * 8 + tg];
      }
      __syncthreads();
    } // l

    // ---- hpn loss on new_h ----
    loss += hpn_head_loss(hcur, tmp, logitsS, pWhpnnl, bhpnnl,
                          Whpn, bhpn, hpn_tgt + t * B_SZ, tid, lane, mt, ng);

    // ---- stop loss: input = [one-hot(hpn_lbl), msg]; regather msg -> tmp ----
    for (int i = tid; i < B_SZ * HID; i += 1024) {
      int b = i >> 9, c = i & 511;
      const int* mi = msg_idx + (t * B_SZ + b) * MAXNB;
      const int* mm = msg_mask + (t * B_SZ + b) * MAXNB;
      float s = 0.f;
      for (int nb = 0; nb < MAXNB; ++nb)
        s += (float)mm[nb] * hbuf[(size_t)mi[nb] * HID + c];
      tmp[i] = s;
    }
    __syncthreads();
    {
      v8f acc[4] = {};
      gemm_ldsA(acc, tmp, pWtoponl, 16, mt, ng, lane);  // msg cols 4..515 packed
#pragma unroll
      for (int j = 0; j < 4; ++j) {
        int n = (ng * 4 + j) * 16 + (lane & 15);
        const float* wr = Wtoponl + (size_t)n * 516;
        float base = btoponl[n];
#pragma unroll
        for (int r = 0; r < 8; ++r) {
          int m = mt * 16 + r + ((lane >> 4) << 3);
          float v = acc[j][r] + base + wr[hpn_lbl[t * B_SZ + m]];  // one-hot tail
          acc[j][r] = v > 0.f ? v : 0.f;
        }
      }
      __syncthreads();                 // all reads of tmp done before aliasing store
      store_tiles(tmp, acc, mt, ng, lane);
    }
    __syncthreads();
    if (tid < B_SZ) {
      float x = btopo[0];
      const float* g = tmp + tid * HID;
      for (int k = 0; k < HID; ++k) x += g[k] * Wtopo[k];
      float tv = (float)stop_tgt[t * B_SZ + tid];
      loss += fmaxf(x, 0.f) - x * tv + log1pf(expf(-fabsf(x)));
    }
    __syncthreads();

    // ---- scatter new_h into slot buffer ----
    for (int i = tid; i < B_SZ * HID; i += 1024) {
      int b = i >> 9, c = i & 511;
      int slot = widx[t * B_SZ + b] + B_SZ;
      hbuf[(size_t)slot * HID + c] = hcur[i];
    }
    __syncthreads();
  } // t

  // ---- final reduction ----
  red[tid] = loss;
  __syncthreads();
  for (int s = 512; s > 0; s >>= 1) {
    if (tid < s) red[tid] += red[tid + s];
    __syncthreads();
  }
  if (tid == 0) out[0] = red[0];
}

// ---------------------------------------------------------------------------
extern "C" void kernel_launch(void* const* d_in, const int* in_sizes, int n_in,
                              void* d_out, int out_size, void* d_ws, size_t ws_size,
                              hipStream_t stream) {
  (void)in_sizes; (void)n_in; (void)out_size; (void)ws_size;
  const float* seq    = (const float*)d_in[0];
  const float* tree   = (const float*)d_in[1];
  const float* graph  = (const float*)d_in[2];
  const int* msg_idx  = (const int*)d_in[3];
  const int* msg_mask = (const int*)d_in[4];
  const int* widx     = (const int*)d_in[5];
  const int* nuc_tgt  = (const int*)d_in[6];
  const int* hpn_tgt  = (const int*)d_in[7];
  const int* stop_tgt = (const int*)d_in[8];
  const int* hpn_lbl  = (const int*)d_in[9];
  const int* root_tgt = (const int*)d_in[10];
  const float* Wz = (const float*)d_in[11];  const float* bz = (const float*)d_in[12];
  const float* Wr = (const float*)d_in[13];  const float* br = (const float*)d_in[14];
  const float* Wh = (const float*)d_in[15];  const float* bh = (const float*)d_in[16];
  const float* Whpn   = (const float*)d_in[17]; const float* bhpn   = (const float*)d_in[18];
  const float* Whpnnl = (const float*)d_in[19]; const float* bhpnnl = (const float*)d_in[20];
  const float* Wnuc   = (const float*)d_in[21]; const float* bnuc   = (const float*)d_in[22];
  const float* Wnucnl = (const float*)d_in[23]; const float* bnucnl = (const float*)d_in[24];
  const float* Wtopo  = (const float*)d_in[25]; const float* btopo  = (const float*)d_in[26];
  const float* Wtoponl= (const float*)d_in[27]; const float* btoponl= (const float*)d_in[28];

  char* ws = (char*)d_ws;
  float* hbuf = (float*)ws;
  size_t off = (size_t)NSLOTS * HID * sizeof(float);      // 8.5 MB
  const size_t pack16 = (size_t)32 * 16 * 32 * 16;        // halves, Kc=16
  const size_t pack24 = (size_t)32 * 24 * 32 * 16;        // halves, Kc=24
  _Float16* pWz     = (_Float16*)(ws + off); off += pack16 * 2;
  _Float16* pWr     = (_Float16*)(ws + off); off += pack16 * 2;
  _Float16* pWh     = (_Float16*)(ws + off); off += pack16 * 2;
  _Float16* pWnucnl = (_Float16*)(ws + off); off += pack24 * 2;
  _Float16* pWhpnnl = (_Float16*)(ws + off); off += pack16 * 2;
  _Float16* pWtoponl= (_Float16*)(ws + off); off += pack16 * 2;

  hipMemsetAsync(hbuf, 0, (size_t)NSLOTS * HID * sizeof(float), stream);

  int t16 = 32 * 16 * 32 * 8, t24 = 32 * 24 * 32 * 8;
  pack_kernel<<<(t16 + 255) / 256, 256, 0, stream>>>(Wz, 517, 5, 16, pWz);
  pack_kernel<<<(t16 + 255) / 256, 256, 0, stream>>>(Wr, 517, 5, 16, pWr);
  pack_kernel<<<(t16 + 255) / 256, 256, 0, stream>>>(Wh, 517, 5, 16, pWh);
  pack_kernel<<<(t24 + 255) / 256, 256, 0, stream>>>(Wnucnl, 768, 0, 24, pWnucnl);
  pack_kernel<<<(t16 + 255) / 256, 256, 0, stream>>>(Whpnnl, 512, 0, 16, pWhpnnl);
  pack_kernel<<<(t16 + 255) / 256, 256, 0, stream>>>(Wtoponl, 516, 4, 16, pWtoponl);

  decoder_kernel<<<1, 1024, 0, stream>>>(
      seq, tree, graph, msg_idx, msg_mask, widx, nuc_tgt, hpn_tgt, stop_tgt,
      hpn_lbl, root_tgt,
      Wz, bz, Wr, br, Wh, bh, Whpn, bhpn, Whpnnl, bhpnnl,
      Wnuc, bnuc, Wnucnl, bnucnl, Wtopo, btopo, Wtoponl, btoponl,
      pWz, pWr, pWh, pWnucnl, pWhpnnl, pWtoponl,
      hbuf, (float*)d_out);
}
